// ResBlock_19945828122641
// MI455X (gfx1250) — matrix operand order repaired and checked by reading, not compiled, not run on hardware
//
#include <hip/hip_runtime.h>
#include <cstdint>

typedef __attribute__((ext_vector_type(16))) __bf16    v16bf;
typedef __attribute__((ext_vector_type(8)))  float     v8f;
typedef __attribute__((ext_vector_type(8)))  unsigned  v8u;
typedef __attribute__((ext_vector_type(4)))  int       v4i;

#define CC 32   // channels (in == out)
#define KK 27   // kernel offsets
#define WSWZ_ELEMS (KK * 1024)          // 27 * 2tiles * 32lanes * 16 = 27648 bf16

__device__ __forceinline__ unsigned short f32_to_bf16_bits(float f) {
  unsigned u = __builtin_bit_cast(unsigned, f);
  u += 0x7fffu + ((u >> 16) & 1u);              // round-to-nearest-even
  return (unsigned short)(u >> 16);
}
__device__ __forceinline__ __bf16 f32_to_bf16(float f) {
  unsigned short h = f32_to_bf16_bits(f);
  return __builtin_bit_cast(__bf16, h);
}
__device__ __forceinline__ float bf16_bits_to_f32(unsigned short h) {
  return __builtin_bit_cast(float, (unsigned)h << 16);
}
__device__ __forceinline__ unsigned pack2_bf16(float lo, float hi) {
  return (unsigned)f32_to_bf16_bits(lo) | ((unsigned)f32_to_bf16_bits(hi) << 16);
}

// One-shot: W[k][kk][c] fp32 -> bf16 in the WMMA B-operand per-lane layout:
// element e = ((k*2 + ctile)*32 + lane)*16 + j holds W[k][(j>>3)*16 + (lane>>4)*8 + (j&7)][ctile*16 + lane%16].
__global__ void swizzle_weights_kernel(const float* __restrict__ W,
                                       __bf16* __restrict__ out) {
  int e = blockIdx.x * 256 + threadIdx.x;
  if (e < WSWZ_ELEMS) {
    int k    = e >> 10;
    int rem  = e & 1023;
    int ct   = rem >> 9;
    int lane = (rem >> 4) & 31;
    int j    = rem & 15;
    int kk   = ((j >> 3) << 4) + ((lane >> 4) << 3) + (j & 7);
    int c    = (ct << 4) + (lane & 15);
    out[e] = f32_to_bf16(W[(k << 10) + kk * CC + c]);
  }
}

// ---------------------------------------------------------------------------
// Fused gather -> WMMA GEMM accumulated over KK offsets. Input rows are bf16;
// gathered bits feed the A operand directly (zero conversion VALU in loop).
// One wave = 16-row x 32-col output tile. Pre-swizzled bf16 weights are
// async-copied into LDS once per block.
// PASS 0 stores bf16 (h1); PASS 1 stores fp32 (h2). Both accumulate
// per-channel sum / sum^2 from the fp32 accumulators for batch-norm.
// ---------------------------------------------------------------------------
template<int PASS>
__global__ __launch_bounds__(256)
void conv_wmma_kernel(const __bf16* __restrict__ Xin, const int* __restrict__ nbr,
                      const __bf16* __restrict__ Wsw,
                      __bf16* __restrict__ Hbf, float* __restrict__ Hf,
                      float* __restrict__ sums, float* __restrict__ sumsq, int N)
{
  __shared__ __align__(32) __bf16 lds_w[WSWZ_ELEMS];   // 54 KB, [k][ctile][lane][16]
  __shared__ float red[2][8][32];

  const int tid = threadIdx.x;

  // Stage pre-swizzled weights (contiguous 54 KB) into LDS via the CDNA5
  // async global->LDS path (ASYNCcnt), falling back to a plain b128 copy.
#if defined(__HIP_DEVICE_COMPILE__) &&                                         \
    __has_builtin(__builtin_amdgcn_global_load_async_to_lds_b128) &&           \
    __has_builtin(__builtin_amdgcn_s_wait_asynccnt)
  {
    typedef __attribute__((address_space(1))) v4i* gv4p;
    typedef __attribute__((address_space(3))) v4i* lv4p;
    gv4p gsrc = (gv4p)Wsw;
    lv4p ldst = (lv4p)lds_w;
    for (int e = tid; e < WSWZ_ELEMS / 8; e += 256)
      __builtin_amdgcn_global_load_async_to_lds_b128(gsrc + e, ldst + e, 0, 0);
    __builtin_amdgcn_s_wait_asynccnt(0);
  }
#else
  for (int e = tid; e < WSWZ_ELEMS / 8; e += 256)
    ((uint4*)lds_w)[e] = ((const uint4*)Wsw)[e];
#endif
  __syncthreads();

  const int wave     = tid >> 5;
  const int lane     = tid & 31;
  const int tilebase = (blockIdx.x * 8 + wave) * 16;
  const int kb       = (lane & 16) ? 8 : 0;   // A-operand K-chunk base per lane

  v8f c0 = {};
  v8f c1 = {};

  if (tilebase < N) {
    const int  r     = lane & 15;
    const bool rowok = (tilebase + r) < N;

    for (int k = 0; k < KK; ++k) {
      int row = 0;
      if (rowok) row = nbr[(size_t)k * N + tilebase + r];

      // Gather this lane's A fragment: channels [kb,kb+8) and [kb+16,kb+24)
      // as two 16B loads of raw bf16 bits (32-bit byte offset, uniform base).
      v8u av = {};
      if (rowok) {
        const unsigned boff = (unsigned)row * (unsigned)(CC * 2) +
                              (unsigned)(kb * 2);
        const uint4* hp = (const uint4*)((const char*)Xin + boff);
        uint4 u0 = hp[0];
        uint4 u1 = hp[2];          // +32B = channels kb+16..kb+23
        av[0] = u0.x; av[1] = u0.y; av[2] = u0.z; av[3] = u0.w;
        av[4] = u1.x; av[5] = u1.y; av[6] = u1.z; av[7] = u1.w;
      }
      v16bf a = __builtin_bit_cast(v16bf, av);

      // B tiles (cols 0-15 and 16-31) from LDS: 16 bf16 (32B) per lane each.
      const v16bf* bw = (const v16bf*)&lds_w[(size_t)(k * 2) * 512 + lane * 16];
      v16bf b0 = bw[0];
      v16bf b1 = bw[32];   // +512 bf16 = next column tile

      c0 = __builtin_amdgcn_wmma_f32_16x16x32_bf16(false, a, false, b0,
                                                   (short)0, c0, false, false);
      c1 = __builtin_amdgcn_wmma_f32_16x16x32_bf16(false, a, false, b1,
                                                   (short)0, c1, false, false);
    }

    // Store tile. C/D layout: lane n%16 = column; VGPR v = row v (+8 hi half).
    const int mhalf = (lane >> 4) << 3;
    const int n0    = lane & 15;
#pragma unroll
    for (int v = 0; v < 8; ++v) {
      int rowo = tilebase + mhalf + v;
      if (rowo < N) {
        if (PASS == 0) {
          Hbf[(size_t)rowo * CC + n0]      = f32_to_bf16(c0[v]);
          Hbf[(size_t)rowo * CC + 16 + n0] = f32_to_bf16(c1[v]);
        } else {
          Hf[(size_t)rowo * CC + n0]      = c0[v];
          Hf[(size_t)rowo * CC + 16 + n0] = c1[v];
        }
      }
    }
  }

  // Per-channel sum / sum-of-squares for batch-norm statistics.
  float s0 = 0.f, q0 = 0.f, s1 = 0.f, q1 = 0.f;
#pragma unroll
  for (int v = 0; v < 8; ++v) {
    s0 += c0[v]; q0 += c0[v] * c0[v];
    s1 += c1[v]; q1 += c1[v] * c1[v];
  }
  s0 += __shfl_xor(s0, 16, 32);  q0 += __shfl_xor(q0, 16, 32);
  s1 += __shfl_xor(s1, 16, 32);  q1 += __shfl_xor(q1, 16, 32);
  if (lane < 16) {
    red[0][wave][lane]      = s0;
    red[0][wave][16 + lane] = s1;
    red[1][wave][lane]      = q0;
    red[1][wave][16 + lane] = q1;
  }
  __syncthreads();
  if (tid < 64) {
    int which = tid >> 5;
    int c     = tid & 31;
    float t = 0.f;
#pragma unroll
    for (int w = 0; w < 8; ++w) t += red[which][w][c];
    atomicAdd((which ? sumsq : sums) + c, t);
  }
}

// x (fp32) -> bf16 table, 8 elements per thread iteration.
__global__ void f32_to_bf16_kernel(const float* __restrict__ in,
                                   __bf16* __restrict__ outb, size_t total) {
  size_t stride = (size_t)gridDim.x * blockDim.x;
  for (size_t i8 = (size_t)blockIdx.x * blockDim.x + threadIdx.x;
       i8 * 8 < total; i8 += stride) {
    size_t i = i8 * 8;
    float4 a = *(const float4*)(in + i);
    float4 b = *(const float4*)(in + i + 4);
    uint4 o;
    o.x = pack2_bf16(a.x, a.y);
    o.y = pack2_bf16(a.z, a.w);
    o.z = pack2_bf16(b.x, b.y);
    o.w = pack2_bf16(b.z, b.w);
    *(uint4*)(outb + i) = o;
  }
}

// h1r = bf16(relu(h1 * scale + shift)) applied ONCE per element (instead of
// 27x inside conv2's gather loop).
__global__ void bn_relu_kernel(const __bf16* __restrict__ h,
                               const float* __restrict__ scale,
                               const float* __restrict__ shift,
                               __bf16* __restrict__ outb, size_t total) {
  __shared__ float sc[CC], sh[CC];
  if (threadIdx.x < CC) {
    sc[threadIdx.x] = scale[threadIdx.x];
    sh[threadIdx.x] = shift[threadIdx.x];
  }
  __syncthreads();
  size_t stride = (size_t)gridDim.x * blockDim.x;
  for (size_t i8 = (size_t)blockIdx.x * blockDim.x + threadIdx.x;
       i8 * 8 < total; i8 += stride) {
    size_t i = i8 * 8;
    int c = (int)(i & (CC - 1));
    uint4 u = *(const uint4*)(h + i);
    unsigned ud[4] = {u.x, u.y, u.z, u.w};
    unsigned od[4];
#pragma unroll
    for (int d = 0; d < 4; ++d) {
      float lo = bf16_bits_to_f32((unsigned short)(ud[d] & 0xffffu));
      float hi = bf16_bits_to_f32((unsigned short)(ud[d] >> 16));
      lo = fmaxf(fmaf(lo, sc[c + 2 * d],     sh[c + 2 * d]),     0.0f);
      hi = fmaxf(fmaf(hi, sc[c + 2 * d + 1], sh[c + 2 * d + 1]), 0.0f);
      od[d] = pack2_bf16(lo, hi);
    }
    uint4 o;
    o.x = od[0]; o.y = od[1]; o.z = od[2]; o.w = od[3];
    *(uint4*)(outb + i) = o;
  }
}

__global__ void zero_stats_kernel(float* stats) {
  if (threadIdx.x < 128) stats[threadIdx.x] = 0.0f;
}

__global__ void finalize_bn_kernel(const float* __restrict__ sums,
                                   const float* __restrict__ sumsq,
                                   const float* __restrict__ gamma,
                                   const float* __restrict__ beta,
                                   float* __restrict__ scale,
                                   float* __restrict__ shift, float invN) {
  int c = threadIdx.x;
  if (c < CC) {
    float m   = sums[c] * invN;
    float var = sumsq[c] * invN - m * m;
    float s   = gamma[c] * rsqrtf(var + 1e-5f);
    scale[c]  = s;
    shift[c]  = beta[c] - m * s;
  }
}

__global__ void epilogue_kernel(const float* __restrict__ H2,
                                const float* __restrict__ X,
                                const float* __restrict__ scale2,
                                const float* __restrict__ shift2,
                                float* __restrict__ out, size_t total) {
  __shared__ float sc[CC], sh[CC];
  if (threadIdx.x < CC) {
    sc[threadIdx.x] = scale2[threadIdx.x];
    sh[threadIdx.x] = shift2[threadIdx.x];
  }
  __syncthreads();
  size_t stride = (size_t)gridDim.x * blockDim.x;
  for (size_t i4 = (size_t)blockIdx.x * blockDim.x + threadIdx.x;
       i4 * 4 < total; i4 += stride) {
    size_t i = i4 * 4;
    float4 h  = *(const float4*)(H2 + i);
    float4 xv = *(const float4*)(X + i);
    int c = (int)(i & (CC - 1));
    float4 o;
    o.x = fmaxf(fmaf(h.x, sc[c + 0], sh[c + 0]) + xv.x, 0.0f);
    o.y = fmaxf(fmaf(h.y, sc[c + 1], sh[c + 1]) + xv.y, 0.0f);
    o.z = fmaxf(fmaf(h.z, sc[c + 2], sh[c + 2]) + xv.z, 0.0f);
    o.w = fmaxf(fmaf(h.w, sc[c + 3], sh[c + 3]) + xv.w, 0.0f);
    *(float4*)(out + i) = o;
  }
}

extern "C" void kernel_launch(void* const* d_in, const int* in_sizes, int n_in,
                              void* d_out, int out_size, void* d_ws, size_t ws_size,
                              hipStream_t stream) {
  const float* x   = (const float*)d_in[0];
  const int*   nbr = (const int*)d_in[1];
  const float* W1  = (const float*)d_in[2];
  const float* g1  = (const float*)d_in[3];
  const float* b1  = (const float*)d_in[4];
  const float* W2  = (const float*)d_in[5];
  const float* g2  = (const float*)d_in[6];
  const float* b2  = (const float*)d_in[7];
  const int N = in_sizes[0] / CC;
  const size_t total = (size_t)N * CC;

  // Workspace (reused):
  //   bufA [0, 64MB): xb (conv1 input), later overwritten with h1r (conv2 input)
  //   h1   [A, A+64MB): conv1 output; dead after bn_relu
  //   h2   [A, A+128MB) fp32: conv2 output, OVERLAYS dead h1
  //   stats (256 floats) + pre-swizzled bf16 weights (2 x 54KB) after h2
  char* ws = (char*)d_ws;
  const size_t A = (total * sizeof(__bf16) + 255) & ~(size_t)255;
  __bf16* bufA = (__bf16*)ws;            // xb, then h1r
  __bf16* h1   = (__bf16*)(ws + A);
  float*  h2   = (float*)(ws + A);       // overlays h1 (h1 dead by then)
  size_t  soff = (A + total * sizeof(float) + 255) & ~(size_t)255;
  float* stats  = (float*)(ws + soff);
  float* sums1  = stats,       *sq1    = stats + 32;
  float* sums2  = stats + 64,  *sq2    = stats + 96;
  float* scale1 = stats + 128, *shift1 = stats + 160;
  float* scale2 = stats + 192, *shift2 = stats + 224;
  __bf16* wswz1 = (__bf16*)(ws + soff + 1024);
  __bf16* wswz2 = wswz1 + WSWZ_ELEMS;

  const int tiles   = (N + 15) / 16;
  const int blocks  = (tiles + 7) / 8;
  const int eblocks = (int)((total / 8 + 255) / 256);
  const int fblocks = (int)((total / 4 + 255) / 256);
  const int wblocks = (WSWZ_ELEMS + 255) / 256;
  const float invN  = 1.0f / (float)N;

  zero_stats_kernel<<<1, 128, 0, stream>>>(stats);
  swizzle_weights_kernel<<<wblocks, 256, 0, stream>>>(W1, wswz1);
  swizzle_weights_kernel<<<wblocks, 256, 0, stream>>>(W2, wswz2);
  f32_to_bf16_kernel<<<eblocks, 256, 0, stream>>>(x, bufA, total);
  conv_wmma_kernel<0><<<blocks, 256, 0, stream>>>(
      bufA, nbr, wswz1, h1, nullptr, sums1, sq1, N);
  finalize_bn_kernel<<<1, 32, 0, stream>>>(sums1, sq1, g1, b1, scale1, shift1, invN);
  bn_relu_kernel<<<eblocks, 256, 0, stream>>>(h1, scale1, shift1, bufA, total);
  conv_wmma_kernel<1><<<blocks, 256, 0, stream>>>(
      bufA, nbr, wswz2, nullptr, h2, sums2, sq2, N);
  finalize_bn_kernel<<<1, 32, 0, stream>>>(sums2, sq2, g2, b2, scale2, shift2, invN);
  epilogue_kernel<<<fblocks, 256, 0, stream>>>(h2, x, scale2, shift2,
                                               (float*)d_out, total);
}
